// RNN_8358006358257
// MI455X (gfx1250) — compile-verified, matrix operand order checked by reading, not run on hardware
//
#include <hip/hip_runtime.h>
#include <math.h>

// Problem constants
#define HSZ   256
#define NOUT  4
#define STEPS 64
#define BATCH 4096
#define BT    16              // batch rows per workgroup
#define NWG   (BATCH / BT)    // 256 workgroups
#define HSTR  257             // padded LDS row stride for h (bank-conflict-free)

// Output buffer offsets (in floats), concatenated in reference return order:
// outputs[64,4096,256], hidden[2,4096,256], probs[64,4096,4], loss[1], inputX[1,4096,4]
#define OFF_OUTPUTS 0L
#define OFF_HIDDEN  67108864L
#define OFF_PROBS   69206016L
#define OFF_LOSS    70254592L
#define OFF_INPUTX  70254593L

#define WELEMS (768 * 256)    // elements per recurrent weight matrix
#define FRAGN  (8 * 32 * 16)  // bf16 elements in one A-fragment buffer (8 KB)

typedef __attribute__((ext_vector_type(16))) __bf16 v16bf;
typedef __attribute__((ext_vector_type(8)))  float  v8f;

__device__ __forceinline__ float sigf(float x) { return 1.0f / (1.0f + __expf(-x)); }

// K-index mapping inside a 16x32 bf16 A fragment (and mirrored B fragment):
// lane half hh (0: lanes 0-15, 1: lanes 16-31), vector element e in [0,16)
__device__ __forceinline__ int kmap(int e, int hh) {
  return (e & 7) + hh * 8 + ((e >= 8) ? 16 : 0);
}

// ---------------------------------------------------------------------------
// Prep: swizzle the three [768,256] fp32 recurrent matrices into bf16 WMMA
// B-fragment layout in workspace. Tile (nt,kt): 32 lanes x 16 bf16 contiguous.
// frag[nt][kt][lane][e] = W[nt*16 + (lane&15)][kt*32 + kmap(e, lane>>4)]
// ---------------------------------------------------------------------------
__global__ void prep_weights_kernel(const float* __restrict__ Whh0,
                                    const float* __restrict__ Wih1,
                                    const float* __restrict__ Whh1,
                                    __bf16* __restrict__ ws) {
  int tid = blockIdx.x * blockDim.x + threadIdx.x;
  if (tid >= WELEMS) return;
  int e    = tid & 15;
  int lane = (tid >> 4) & 31;
  int tile = tid >> 9;             // 512 elements per tile
  int kt = tile & 7;               // K chunk (0..7)
  int nt = tile >> 3;              // N chunk (0..47)
  int n = nt * 16 + (lane & 15);
  int k = kt * 32 + kmap(e, lane >> 4);
  int src = n * 256 + k;           // W is [768,256] row-major
  ws[tid]              = (__bf16)Whh0[src];
  ws[WELEMS + tid]     = (__bf16)Wih1[src];
  ws[2 * WELEMS + tid] = (__bf16)Whh1[src];
}

// Zero the loss slot and the inputX output (all-zeros tensor).
__global__ void init_out_kernel(float* __restrict__ out) {
  int i = blockIdx.x * blockDim.x + threadIdx.x;
  if (i < BATCH * NOUT) out[OFF_INPUTX + i] = 0.0f;
  if (i == 0) out[OFF_LOSS] = 0.0f;
}

// Cooperative A-fragment build: all 8 waves share the same 16 batch rows, so
// the 8 KB fragment buffer is built once by the whole workgroup. Thread t
// builds the 32-byte slot for (kt = t>>5, lane = t&31).
__device__ __forceinline__ void build_frags_lds(__bf16* frag, const float* hs,
                                                int t) {
  const int kt = t >> 5;
  const int ln = t & 31;
  const int m  = ln & 15;
  const int hh = ln >> 4;
  const float* base = hs + m * HSTR + kt * 32 + hh * 8;
  v16bf a;
#pragma unroll
  for (int e = 0; e < 16; ++e) {
    a[e] = (__bf16)base[(e & 7) + ((e >= 8) ? 16 : 0)];
  }
  *(v16bf*)&frag[(size_t)t * 16] = a;
}

// One 16x16 output tile: accumulate 8 WMMAs over K=256.
// A streamed from the shared LDS fragment buffer, B streamed from global ws.
__device__ __forceinline__ v8f mmtile(const __bf16* frag, const __bf16* wbase,
                                      int nt, int lane) {
  v8f c = {};
  const __bf16* wp = wbase + (size_t)nt * 8 * 512 + lane * 16;
#pragma unroll
  for (int kt = 0; kt < 8; ++kt) {
    v16bf a = *(const v16bf*)(frag + ((size_t)kt * 32 + lane) * 16);
    v16bf b = *(const v16bf*)(wp + kt * 512);
    c = __builtin_amdgcn_wmma_f32_16x16x32_bf16(false, a, false, b,
                                                (short)0, c, false, false);
  }
  return c;
}

// ---------------------------------------------------------------------------
// Main persistent kernel: each workgroup owns BT=16 batch rows and runs the
// full 64-step, 2-layer GRU scan with h0/h1 resident in LDS.
// Wave w owns hidden-unit chunks uc in {2w, 2w+1}; gate tiles are
// nt = uc (r), 16+uc (z), 32+uc (n). Gates computed sequentially (r -> z -> n)
// so at most 2 accumulators + r[8]/z[8] scalars are live -> no spills.
// ---------------------------------------------------------------------------
__launch_bounds__(256)
__global__ void gru_scan_kernel(const float* __restrict__ inputs,
                                const float* __restrict__ Wih0g,   // [768,4]
                                const float* __restrict__ bih0g,
                                const float* __restrict__ bhh0g,
                                const float* __restrict__ bih1g,
                                const float* __restrict__ bhh1g,
                                const float* __restrict__ Woutg,   // [4,256]
                                const float* __restrict__ boutg,
                                const __bf16* __restrict__ ws,
                                float* __restrict__ out) {
  __shared__ float  h0s[BT * HSTR];
  __shared__ float  h1s[BT * HSTR];
  __shared__ __align__(32) __bf16 fragA[FRAGN];   // A-frags: h0 / h0n
  __shared__ __align__(32) __bf16 fragB[FRAGN];   // A-frags: h1
  __shared__ float4 wih0s[768];     // W_ih0 row j as float4
  __shared__ float  bih0s[768], bhh0s[768], bih1s[768], bhh1s[768];
  __shared__ float  wouts[4 * 256];
  __shared__ float  bouts[4];
  __shared__ float4 xts4[BT];       // x_t tile, one float4 per row
  __shared__ float  logits_s[BT * 4];

  const int t    = threadIdx.x;
  const int lane = t & 31;
  const int wave = t >> 5;          // 0..7
  const int hh   = lane >> 4;       // M half select for C rows
  const int nl   = lane & 15;       // N within tile
  const int wgBase = blockIdx.x * BT;

  // Preamble: small weights/biases to LDS, zero h state.
  for (int i = t; i < 768; i += 256) {
    wih0s[i] = make_float4(Wih0g[i * 4 + 0], Wih0g[i * 4 + 1],
                           Wih0g[i * 4 + 2], Wih0g[i * 4 + 3]);
    bih0s[i] = bih0g[i];
    bhh0s[i] = bhh0g[i];
    bih1s[i] = bih1g[i];
    bhh1s[i] = bhh1g[i];
  }
  for (int i = t; i < 1024; i += 256) wouts[i] = Woutg[i];
  if (t < 4) bouts[t] = boutg[t];
  for (int b = 0; b < BT; ++b) {
    h0s[b * HSTR + t] = 0.0f;
    h1s[b * HSTR + t] = 0.0f;
  }
  __syncthreads();

  // Always-zero offset, opaque to the optimizer: keeps the weight-fragment
  // load addresses loop-variant (so LICM cannot preload 64 steps' worth of
  // B-tiles into registers and spill), while the pointers still derive from
  // the kernel argument `ws` -> global address space -> global_load_b128.
  size_t opaq = 0;

  for (int step = 0; step < STEPS; ++step) {
    asm volatile("" : "+s"(opaq));
    const __bf16* wHH0 = ws + opaq;
    const __bf16* wIH1 = ws + WELEMS + opaq;
    const __bf16* wHH1 = ws + 2 * WELEMS + opaq;

    // x_t = inputs[step-1] (zeros at step 0)
    if (t < BT) {
      if (step == 0) {
        xts4[t] = make_float4(0.f, 0.f, 0.f, 0.f);
      } else {
        xts4[t] = ((const float4*)inputs)[(size_t)(step - 1) * BATCH + wgBase + t];
      }
    }
    build_frags_lds(fragA, h0s, t);     // fragments of old h0
    __syncthreads();

    // ---- Layer 0: gh0 = h0 @ Whh0^T (WMMA); gi0 = x @ Wih0^T (VALU, K=4) ----
#pragma unroll 1
    for (int j = 0; j < 2; ++j) {
      const int uc = 2 * wave + j;
      const int u  = uc * 16 + nl;
      float rr[8], zz[8];
      {
        v8f acc = mmtile(fragA, wHH0, uc, lane);
        const float4 wr = wih0s[u];
        const float bir = bih0s[u], bhr = bhh0s[u];
#pragma unroll
        for (int v = 0; v < 8; ++v) {
          const float4 x = xts4[v + 8 * hh];
          float gir = bir + x.x * wr.x + x.y * wr.y + x.z * wr.z + x.w * wr.w;
          rr[v] = sigf(gir + acc[v] + bhr);
        }
      }
      {
        v8f acc = mmtile(fragA, wHH0, 16 + uc, lane);
        const float4 wz = wih0s[256 + u];
        const float biz = bih0s[256 + u], bhz = bhh0s[256 + u];
#pragma unroll
        for (int v = 0; v < 8; ++v) {
          const float4 x = xts4[v + 8 * hh];
          float giz = biz + x.x * wz.x + x.y * wz.y + x.z * wz.z + x.w * wz.w;
          zz[v] = sigf(giz + acc[v] + bhz);
        }
      }
      {
        v8f acc = mmtile(fragA, wHH0, 32 + uc, lane);
        const float4 wn = wih0s[512 + u];
        const float bin = bih0s[512 + u], bhn = bhh0s[512 + u];
#pragma unroll
        for (int v = 0; v < 8; ++v) {
          const int b = v + 8 * hh;
          const float4 x = xts4[b];
          float gin = bin + x.x * wn.x + x.y * wn.y + x.z * wn.z + x.w * wn.w;
          float nn = tanhf(gin + rr[v] * (acc[v] + bhn));
          float h  = h0s[b * HSTR + u];
          h0s[b * HSTR + u] = (1.0f - zz[v]) * nn + zz[v] * h;   // h0n
        }
      }
    }
    __syncthreads();

    // ---- Layer 1: gi1 = h0n @ Wih1^T, gh1 = h1 @ Whh1^T ----
    build_frags_lds(fragA, h0s, t);     // fragments of h0n
    build_frags_lds(fragB, h1s, t);     // fragments of old h1
    __syncthreads();
#pragma unroll 1
    for (int j = 0; j < 2; ++j) {
      const int uc = 2 * wave + j;
      const int u  = uc * 16 + nl;
      float rr[8], zz[8];
      {
        v8f ci = mmtile(fragA, wIH1, uc, lane);
        v8f ch = mmtile(fragB, wHH1, uc, lane);
        const float bir = bih1s[u], bhr = bhh1s[u];
#pragma unroll
        for (int v = 0; v < 8; ++v)
          rr[v] = sigf(ci[v] + bir + ch[v] + bhr);
      }
      {
        v8f ci = mmtile(fragA, wIH1, 16 + uc, lane);
        v8f ch = mmtile(fragB, wHH1, 16 + uc, lane);
        const float biz = bih1s[256 + u], bhz = bhh1s[256 + u];
#pragma unroll
        for (int v = 0; v < 8; ++v)
          zz[v] = sigf(ci[v] + biz + ch[v] + bhz);
      }
      {
        v8f ci = mmtile(fragA, wIH1, 32 + uc, lane);
        v8f ch = mmtile(fragB, wHH1, 32 + uc, lane);
        const float bin = bih1s[512 + u], bhn = bhh1s[512 + u];
#pragma unroll
        for (int v = 0; v < 8; ++v) {
          const int b = v + 8 * hh;
          float nn = tanhf(ci[v] + bin + rr[v] * (ch[v] + bhn));
          float h  = h1s[b * HSTR + u];
          float hn = (1.0f - zz[v]) * nn + zz[v] * h;
          h1s[b * HSTR + u] = hn;
          // outputs is a 268 MB write-once stream: non-temporal store
          __builtin_nontemporal_store(
              hn, &out[OFF_OUTPUTS + (size_t)step * BATCH * HSZ +
                       (size_t)(wgBase + b) * HSZ + u]);
        }
      }
    }
    __syncthreads();

    // ---- Output projection (H=256 -> 4), log-softmax, probs, loss ----
    if (t < BT * 4) {
      const int b = t >> 2, c = t & 3;
      float acc = bouts[c];
      for (int u2 = 0; u2 < 256; ++u2) acc += h1s[b * HSTR + u2] * wouts[c * 256 + u2];
      logits_s[t] = acc;
    }
    __syncthreads();
    if (t < BT) {
      const float l0 = logits_s[t * 4 + 0], l1 = logits_s[t * 4 + 1];
      const float l2 = logits_s[t * 4 + 2], l3 = logits_s[t * 4 + 3];
      const float m  = fmaxf(fmaxf(l0, l1), fmaxf(l2, l3));
      const float e0 = __expf(l0 - m), e1 = __expf(l1 - m);
      const float e2 = __expf(l2 - m), e3 = __expf(l3 - m);
      const float s = e0 + e1 + e2 + e3;
      const float inv = 1.0f / s;
      float* pb = &out[OFF_PROBS + (size_t)step * BATCH * NOUT +
                       (size_t)(wgBase + t) * NOUT];
      __builtin_nontemporal_store(e0 * inv, pb + 0);
      __builtin_nontemporal_store(e1 * inv, pb + 1);
      __builtin_nontemporal_store(e2 * inv, pb + 2);
      __builtin_nontemporal_store(e3 * inv, pb + 3);
      // target = argmax(inputs[step][row]) (first max wins, like jnp.argmax)
      const float* xin = inputs + (size_t)step * BATCH * NOUT +
                         (size_t)(wgBase + t) * NOUT;
      float best = xin[0]; int tg = 0;
      for (int c2 = 1; c2 < 4; ++c2) {
        float vx = xin[c2];
        if (vx > best) { best = vx; tg = c2; }
      }
      const float lse = m + __logf(s);
      const float lt  = (tg == 0) ? l0 : (tg == 1) ? l1 : (tg == 2) ? l2 : l3;
      atomicAdd(&out[OFF_LOSS], -(lt - lse) * (1.0f / BATCH));
    }
    __syncthreads();
  }

  // Final hidden state [2, B, H]
  for (int b = 0; b < BT; ++b) {
    out[OFF_HIDDEN + (size_t)(wgBase + b) * HSZ + t] = h0s[b * HSTR + t];
    out[OFF_HIDDEN + (size_t)BATCH * HSZ + (size_t)(wgBase + b) * HSZ + t] =
        h1s[b * HSTR + t];
  }
}

extern "C" void kernel_launch(void* const* d_in, const int* in_sizes, int n_in,
                              void* d_out, int out_size, void* d_ws, size_t ws_size,
                              hipStream_t stream) {
  const float* inputs = (const float*)d_in[0];
  const float* Wih0   = (const float*)d_in[1];
  const float* Whh0   = (const float*)d_in[2];
  const float* bih0   = (const float*)d_in[3];
  const float* bhh0   = (const float*)d_in[4];
  const float* Wih1   = (const float*)d_in[5];
  const float* Whh1   = (const float*)d_in[6];
  const float* bih1   = (const float*)d_in[7];
  const float* bhh1   = (const float*)d_in[8];
  const float* Wout   = (const float*)d_in[9];
  const float* bout   = (const float*)d_in[10];
  float*  out = (float*)d_out;
  __bf16* ws  = (__bf16*)d_ws;   // needs 3*768*256*2 = 1.13 MB

  prep_weights_kernel<<<(WELEMS + 255) / 256, 256, 0, stream>>>(Whh0, Wih1, Whh1, ws);
  init_out_kernel<<<(BATCH * NOUT + 255) / 256, 256, 0, stream>>>(out);
  gru_scan_kernel<<<NWG, 256, 0, stream>>>(inputs, Wih0, bih0, bhh0, bih1, bhh1,
                                           Wout, bout, ws, out);
}
